// AtomicScaleModule_26405458935844
// MI455X (gfx1250) — matrix operand order repaired and checked by reading, not compiled, not run on hardware
//
#include <hip/hip_runtime.h>
#include <hip/hip_bf16.h>

typedef __attribute__((ext_vector_type(16))) _Float16 v16h;
typedef __attribute__((ext_vector_type(8)))  float    v8f;

#define AFL 64
#define KNBR 12

__device__ __forceinline__ void wave_fence() {
  asm volatile("s_wait_dscnt 0" ::: "memory");
}

__device__ __forceinline__ v8f wmma16(v16h a, v16h b, v8f c) {
  return __builtin_amdgcn_wmma_f32_16x16x32_f16(false, a, false, b, (short)0, c, false, false);
}

// ================= fragment helpers (layouts per CDNA5 ISA §7.12.2) =================

// --- B fragments: weights pre-swizzled so one lane's 16 f16 are contiguous (2x ds_load_b128) ---
// B 32x16 f16: lane = ((kk>>3)&1)*16 + nn ; elem = ((kk>>4)&1)*8 + (kk&7)  (kk = k%32, nn = n%16)
__device__ __forceinline__ void store_b_swz(_Float16* dst, int NT, int k, int n, float v) {
  int kc = k >> 5, kk = k & 31;
  int nt = n >> 4, nn = n & 15;
  int lane = (((kk >> 3) & 1) << 4) + nn;
  int elem = (((kk >> 4) & 1) << 3) + (kk & 7);
  dst[(((kc * NT + nt) * 32 + lane) << 4) + elem] = (_Float16)v;
}

__device__ __forceinline__ v16h load_b_swz(const _Float16* base, int NT, int kc, int nt) {
  unsigned lane = threadIdx.x & 31u;
  return *(const v16h*)(base + (((kc * NT + nt) * 32 + lane) << 4));
}

// --- A-swizzled LDS scratch for D->A layout crossing (hidden/agg activations) ---
// A 16x32 f16: lane = ((kk>>3)&1)*16 + m ; elem = (((kk>>4)&1)*4 | ((kk>>1)&3))*2 | (kk&1)
__device__ __forceinline__ int a_swz_off(int m, int k_abs) {
  int kc = k_abs >> 5, kk = k_abs & 31;
  int lane = (((kk >> 3) & 1) << 4) + m;
  int j = (((kk >> 4) & 1) << 2) | ((kk >> 1) & 3);
  int elem = (j << 1) | (kk & 1);
  return ((kc * 32 + lane) << 4) + elem;
}

__device__ __forceinline__ v16h load_a_swz(const _Float16* base, int kc) {
  unsigned lane = threadIdx.x & 31u;
  return *(const v16h*)(base + ((kc * 32 + lane) << 4));
}

// --- A fragment built directly from a row-major f32 global row (one atom row per lane) ---
__device__ __forceinline__ v16h make_a_frag(const float* rowp, int c0) {
  unsigned h = (threadIdx.x & 31u) >> 4;
  v16h a;
#pragma unroll
  for (int j = 0; j < 8; ++j) {
    int kb = ((j & 4) << 2) + (int)(h << 3) + ((j & 3) << 1);
    float2 p = *(const float2*)(rowp + c0 + kb);
    a[2 * j]     = (_Float16)p.x;
    a[2 * j + 1] = (_Float16)p.y;
  }
  return a;
}

// same, but zero-fills columns >= lim (for the 92->96 padded embedding)
__device__ __forceinline__ v16h make_a_frag_lim(const float* rowp, int c0, int lim) {
  unsigned h = (threadIdx.x & 31u) >> 4;
  v16h a;
#pragma unroll
  for (int j = 0; j < 8; ++j) {
    int kb = ((j & 4) << 2) + (int)(h << 3) + ((j & 3) << 1);
    int c = c0 + kb;
    if (c < lim) {
      float2 p = *(const float2*)(rowp + c);
      a[2 * j]     = (_Float16)p.x;
      a[2 * j + 1] = (_Float16)p.y;
    } else {
      a[2 * j]     = (_Float16)0.f;
      a[2 * j + 1] = (_Float16)0.f;
    }
  }
  return a;
}

// ============================ embedding: [N,92] @ [92,64] + b ============================
__global__ void cg_embed_kernel(const float* __restrict__ atom_fea,
                                const float* __restrict__ ae_w,
                                const float* __restrict__ ae_b,
                                float* __restrict__ a_out,
                                int n_tiles, int N) {
  __shared__ __align__(32) _Float16 s_w[3 * 4 * 512];   // 12 KB, B-swizzled, K padded to 96
  int tid = threadIdx.x;
  for (int i = tid; i < 96 * 64; i += blockDim.x) {
    int k = i >> 6, n = i & 63;
    store_b_swz(s_w, 4, k, n, (k < 92) ? ae_w[k * 64 + n] : 0.f);
  }
  __syncthreads();

  int wave = tid >> 5;
  int tile = blockIdx.x * (blockDim.x >> 5) + wave;
  if (tile >= n_tiles) return;
  int base = tile * 16;
  if (base + 16 > N) base = N - 16;
  unsigned lane = tid & 31u, h = lane >> 4, nn = lane & 15u;

  const float* rowp = atom_fea + (size_t)(base + nn) * 92;
  v16h a0 = make_a_frag(rowp, 0);
  v16h a1 = make_a_frag(rowp, 32);
  v16h a2 = make_a_frag_lim(rowp, 64, 92);

#pragma unroll
  for (int nt = 0; nt < 4; ++nt) {
    v8f c = {};
    c = wmma16(a0, load_b_swz(s_w, 4, 0, nt), c);
    c = wmma16(a1, load_b_swz(s_w, 4, 1, nt), c);
    c = wmma16(a2, load_b_swz(s_w, 4, 2, nt), c);
    float bias = ae_b[nt * 16 + nn];
#pragma unroll
    for (int r = 0; r < 8; ++r)
      a_out[(size_t)(base + r + (int)h * 8) * AFL + nt * 16 + nn] = c[r] + bias;
  }
}

// ============================ one conv layer (message + update) ============================
__global__ void cg_conv_kernel(const float* __restrict__ a_in,
                               const int* __restrict__ nbr_idx,
                               const float* __restrict__ w1m, const float* __restrict__ b1m,
                               const float* __restrict__ w2m, const float* __restrict__ b2m,
                               const float* __restrict__ w1u, const float* __restrict__ b1u,
                               const float* __restrict__ w2u, const float* __restrict__ b2u,
                               float* __restrict__ a_out,
                               int n_tiles, int N) {
  __shared__ __align__(32) _Float16 s_w1m[4 * 4 * 512];   // 16 KB  (128x64, B-swizzled)
  __shared__ __align__(32) _Float16 s_w2m[2 * 4 * 512];   //  8 KB  (64x64)
  __shared__ __align__(32) _Float16 s_w1u[4 * 4 * 512];   // 16 KB
  __shared__ __align__(32) _Float16 s_w2u[2 * 4 * 512];   //  8 KB
  __shared__ __align__(32) _Float16 s_swz[4][2 * 1024];   // 16 KB (per-wave: two 16x64 A-swz tiles)
  int tid = threadIdx.x;
  for (int i = tid; i < 128 * 64; i += blockDim.x) {
    int k = i >> 6, n = i & 63;
    store_b_swz(s_w1m, 4, k, n, w1m[i]);
    store_b_swz(s_w1u, 4, k, n, w1u[i]);
  }
  for (int i = tid; i < 64 * 64; i += blockDim.x) {
    int k = i >> 6, n = i & 63;
    store_b_swz(s_w2m, 4, k, n, w2m[i]);
    store_b_swz(s_w2u, 4, k, n, w2u[i]);
  }
  __syncthreads();

  int wave = tid >> 5;
  int tile = blockIdx.x * (blockDim.x >> 5) + wave;
  if (tile >= n_tiles) return;
  int base = tile * 16;
  if (base + 16 > N) base = N - 16;
  unsigned lane = tid & 31u, h = lane >> 4, nn = lane & 15u;
  _Float16* swz = s_swz[wave];

  int ai = base + (int)nn;                       // this lane's atom row
  const float* selfrow = a_in + (size_t)ai * AFL;
  v16h ea0 = make_a_frag(selfrow, 0);            // invariant over neighbors
  v16h ea1 = make_a_frag(selfrow, 32);

  float bias1[4], bias2[4];
#pragma unroll
  for (int nt = 0; nt < 4; ++nt) {
    bias1[nt] = b1m[nt * 16 + nn];
    bias2[nt] = 12.f * b2m[nt * 16 + nn];
  }

  const int2* idx2 = (const int2*)(nbr_idx + (size_t)ai * KNBR);  // 8B-aligned (stride 48B)
  v8f agg[4] = {v8f{}, v8f{}, v8f{}, v8f{}};

  for (int p = 0; p < KNBR / 2; ++p) {
    int2 ij = idx2[p];
    const float* rowA = a_in + (size_t)ij.x * AFL;
    const float* rowB = a_in + (size_t)ij.y * AFL;
    if (p + 1 < KNBR / 2) {                      // warm L0 for next pair's gather rows
      int2 nx = idx2[p + 1];
      __builtin_prefetch(a_in + (size_t)nx.x * AFL, 0, 0);
      __builtin_prefetch(a_in + (size_t)nx.y * AFL, 0, 0);
    }
    v16h eaA2 = make_a_frag(rowA, 0), eaA3 = make_a_frag(rowA, 32);
    v16h eaB2 = make_a_frag(rowB, 0), eaB3 = make_a_frag(rowB, 32);

    // message layer 1 for both edges, sharing each B fragment across two WMMAs
#pragma unroll
    for (int nt = 0; nt < 4; ++nt) {
      v8f ca = {}, cb = {};
      v16h b0 = load_b_swz(s_w1m, 4, 0, nt);
      ca = wmma16(ea0, b0, ca);  cb = wmma16(ea0, b0, cb);
      v16h b1 = load_b_swz(s_w1m, 4, 1, nt);
      ca = wmma16(ea1, b1, ca);  cb = wmma16(ea1, b1, cb);
      v16h b2 = load_b_swz(s_w1m, 4, 2, nt);
      ca = wmma16(eaA2, b2, ca); cb = wmma16(eaB2, b2, cb);
      v16h b3 = load_b_swz(s_w1m, 4, 3, nt);
      ca = wmma16(eaA3, b3, ca); cb = wmma16(eaB3, b3, cb);
#pragma unroll
      for (int r = 0; r < 8; ++r) {
        int o = a_swz_off(r + (int)h * 8, nt * 16 + nn);
        swz[o]        = (_Float16)fmaxf(ca[r] + bias1[nt], 0.f);
        swz[o + 1024] = (_Float16)fmaxf(cb[r] + bias1[nt], 0.f);
      }
    }
    wave_fence();

    // message layer 2: both hiddens against each B fragment; sum through WMMA C operand
    v16h haA0 = load_a_swz(swz, 0),        haA1 = load_a_swz(swz, 1);
    v16h haB0 = load_a_swz(swz + 1024, 0), haB1 = load_a_swz(swz + 1024, 1);
    wave_fence();
#pragma unroll
    for (int nt = 0; nt < 4; ++nt) {
      v16h b0 = load_b_swz(s_w2m, 4, 0, nt);
      agg[nt] = wmma16(haA0, b0, agg[nt]);
      agg[nt] = wmma16(haB0, b0, agg[nt]);
      v16h b1 = load_b_swz(s_w2m, 4, 1, nt);
      agg[nt] = wmma16(haA1, b1, agg[nt]);
      agg[nt] = wmma16(haB1, b1, agg[nt]);
    }
  }

  // u_in = [a_self | agg + 12*b2m]; second half goes through A-swizzled LDS
#pragma unroll
  for (int nt = 0; nt < 4; ++nt) {
#pragma unroll
    for (int r = 0; r < 8; ++r)
      swz[a_swz_off(r + (int)h * 8, nt * 16 + nn)] = (_Float16)(agg[nt][r] + bias2[nt]);
  }
  wave_fence();
  v16h ua2 = load_a_swz(swz, 0);
  v16h ua3 = load_a_swz(swz, 1);
  wave_fence();

  // update layer 1: 128 -> 64, relu
#pragma unroll
  for (int nt = 0; nt < 4; ++nt) {
    v8f c = {};
    c = wmma16(ea0, load_b_swz(s_w1u, 4, 0, nt), c);
    c = wmma16(ea1, load_b_swz(s_w1u, 4, 1, nt), c);
    c = wmma16(ua2, load_b_swz(s_w1u, 4, 2, nt), c);
    c = wmma16(ua3, load_b_swz(s_w1u, 4, 3, nt), c);
    float bias = b1u[nt * 16 + nn];
#pragma unroll
    for (int r = 0; r < 8; ++r)
      swz[a_swz_off(r + (int)h * 8, nt * 16 + nn)] = (_Float16)fmaxf(c[r] + bias, 0.f);
  }
  wave_fence();

  // update layer 2: 64 -> 64, residual + relu -> a_out (no per-element guards: base clamped)
  v16h hb0 = load_a_swz(swz, 0);
  v16h hb1 = load_a_swz(swz, 1);
#pragma unroll
  for (int nt = 0; nt < 4; ++nt) {
    v8f c = {};
    c = wmma16(hb0, load_b_swz(s_w2u, 4, 0, nt), c);
    c = wmma16(hb1, load_b_swz(s_w2u, 4, 1, nt), c);
    float bias = b2u[nt * 16 + nn];
#pragma unroll
    for (int r = 0; r < 8; ++r) {
      size_t off = (size_t)(base + r + (int)h * 8) * AFL + nt * 16 + nn;
      a_out[off] = fmaxf(a_in[off] + c[r] + bias, 0.f);
    }
  }
}

// ============================ head: fx MLP + property projection ============================
__global__ void cg_head_kernel(const float* __restrict__ a_in,
                               const float* __restrict__ fxw1, const float* __restrict__ fxb1,
                               const float* __restrict__ fxw2, const float* __restrict__ fxb2,
                               const float* __restrict__ pw,   const float* __restrict__ pb,
                               float* __restrict__ props, float* __restrict__ af,
                               int n_tiles, int N) {
  __shared__ __align__(32) _Float16 s_w1[2 * 8 * 512];   // 16 KB  (64x128, B-swizzled, NT=8)
  __shared__ __align__(32) _Float16 s_w2[4 * 4 * 512];   // 16 KB  (128x64, NT=4)
  __shared__ __align__(32) _Float16 s_swz[4][4 * 512];   // 16 KB  (per-wave, 16x128 hidden)
  int tid = threadIdx.x;
  for (int i = tid; i < 64 * 128; i += blockDim.x) {
    store_b_swz(s_w1, 8, i >> 7, i & 127, fxw1[i]);
    store_b_swz(s_w2, 4, i >> 6, i & 63, fxw2[i]);
  }
  __syncthreads();

  int wave = tid >> 5;
  int tile = blockIdx.x * (blockDim.x >> 5) + wave;
  if (tile >= n_tiles) return;
  int base = tile * 16;
  if (base + 16 > N) base = N - 16;
  unsigned lane = tid & 31u, h = lane >> 4, nn = lane & 15u;
  _Float16* swz = s_swz[wave];

  const float* rowp = a_in + (size_t)(base + nn) * AFL;
  v16h aa0 = make_a_frag(rowp, 0);
  v16h aa1 = make_a_frag(rowp, 32);

  // fx layer 1: 64 -> 128, relu -> A-swizzled LDS
#pragma unroll
  for (int nt = 0; nt < 8; ++nt) {
    v8f c = {};
    c = wmma16(aa0, load_b_swz(s_w1, 8, 0, nt), c);
    c = wmma16(aa1, load_b_swz(s_w1, 8, 1, nt), c);
    float bias = fxb1[nt * 16 + nn];
#pragma unroll
    for (int r = 0; r < 8; ++r)
      swz[a_swz_off(r + (int)h * 8, nt * 16 + nn)] = (_Float16)fmaxf(c[r] + bias, 0.f);
  }
  wave_fence();

  // fx layer 2: 128 -> 64 -> af  (results kept in regs for the property projection)
  v16h hb[4];
#pragma unroll
  for (int kc = 0; kc < 4; ++kc) hb[kc] = load_a_swz(swz, kc);
  v8f afv[4];
#pragma unroll
  for (int nt = 0; nt < 4; ++nt) {
    v8f c = {};
#pragma unroll
    for (int kc = 0; kc < 4; ++kc)
      c = wmma16(hb[kc], load_b_swz(s_w2, 4, kc, nt), c);
    float bias = fxb2[nt * 16 + nn];
#pragma unroll
    for (int r = 0; r < 8; ++r) {
      c[r] += bias;
      af[(size_t)(base + r + (int)h * 8) * AFL + nt * 16 + nn] = c[r];
    }
    afv[nt] = c;
  }

  // props[i][p] = sum_n af[i][n]*pw[p][n] + pb[p]; cross-lane reduce over the 16-lane half
  float pwv[4][4];
#pragma unroll
  for (int p = 0; p < 4; ++p)
#pragma unroll
    for (int nt = 0; nt < 4; ++nt)
      pwv[p][nt] = pw[p * 64 + nt * 16 + nn];

#pragma unroll
  for (int p = 0; p < 4; ++p) {
    float part[8];
#pragma unroll
    for (int r = 0; r < 8; ++r) {
      part[r] = afv[0][r] * pwv[p][0] + afv[1][r] * pwv[p][1] +
                afv[2][r] * pwv[p][2] + afv[3][r] * pwv[p][3];
#pragma unroll
      for (int mask = 1; mask <= 8; mask <<= 1)
        part[r] += __shfl_xor(part[r], mask, 32);
    }
    if (((int)nn & 3) == p) {
      int r0 = (int)nn >> 2;   // 0..3
      props[(size_t)(base + (int)h * 8 + r0) * 4 + p]     = part[r0] + pb[p];
      props[(size_t)(base + (int)h * 8 + r0 + 4) * 4 + p] = part[r0 + 4] + pb[p];
    }
  }
}

extern "C" void kernel_launch(void* const* d_in, const int* in_sizes, int n_in,
                              void* d_out, int out_size, void* d_ws, size_t ws_size,
                              hipStream_t stream) {
  (void)n_in; (void)out_size; (void)ws_size;
  const float* atom_fea = (const float*)d_in[0];
  // d_in[1] nbr_fea and d_in[5..6] ne_w/ne_b are dead in the reference graph
  const int*   nbr_idx  = (const int*)d_in[2];
  const float* ae_w   = (const float*)d_in[3];
  const float* ae_b   = (const float*)d_in[4];
  const float* msg_w1 = (const float*)d_in[7];
  const float* msg_b1 = (const float*)d_in[8];
  const float* msg_w2 = (const float*)d_in[9];
  const float* msg_b2 = (const float*)d_in[10];
  const float* upd_w1 = (const float*)d_in[11];
  const float* upd_b1 = (const float*)d_in[12];
  const float* upd_w2 = (const float*)d_in[13];
  const float* upd_b2 = (const float*)d_in[14];
  const float* fx_w1  = (const float*)d_in[15];
  const float* fx_b1  = (const float*)d_in[16];
  const float* fx_w2  = (const float*)d_in[17];
  const float* fx_b2  = (const float*)d_in[18];
  const float* pw     = (const float*)d_in[19];
  const float* pb     = (const float*)d_in[20];

  const int N = in_sizes[0] / 92;
  const int n_tiles = (N + 15) / 16;

  float* props = (float*)d_out;                 // [N,4]
  float* af    = props + (size_t)N * 4;         // [N,64]
  float* raw   = af + (size_t)N * AFL;          // [N,64]
  float* bufA  = (float*)d_ws;                  // [N,64] ping-pong scratch

  const int wpb = 4;                            // waves per block (4 x 16-atom tiles)
  const int blocks = (n_tiles + wpb - 1) / wpb;
  dim3 blk(32 * wpb);

  cg_embed_kernel<<<blocks, blk, 0, stream>>>(atom_fea, ae_w, ae_b, bufA, n_tiles, N);

  // conv 0: bufA -> raw
  cg_conv_kernel<<<blocks, blk, 0, stream>>>(bufA, nbr_idx,
      msg_w1 + 0 * 128 * 64, msg_b1 + 0 * 64, msg_w2 + 0 * 64 * 64, msg_b2 + 0 * 64,
      upd_w1 + 0 * 128 * 64, upd_b1 + 0 * 64, upd_w2 + 0 * 64 * 64, upd_b2 + 0 * 64,
      raw, n_tiles, N);
  // conv 1: raw -> bufA
  cg_conv_kernel<<<blocks, blk, 0, stream>>>(raw, nbr_idx,
      msg_w1 + 1 * 128 * 64, msg_b1 + 1 * 64, msg_w2 + 1 * 64 * 64, msg_b2 + 1 * 64,
      upd_w1 + 1 * 128 * 64, upd_b1 + 1 * 64, upd_w2 + 1 * 64 * 64, upd_b2 + 1 * 64,
      bufA, n_tiles, N);
  // conv 2: bufA -> raw   (leaves final `raw` output in place)
  cg_conv_kernel<<<blocks, blk, 0, stream>>>(bufA, nbr_idx,
      msg_w1 + 2 * 128 * 64, msg_b1 + 2 * 64, msg_w2 + 2 * 64 * 64, msg_b2 + 2 * 64,
      upd_w1 + 2 * 128 * 64, upd_b1 + 2 * 64, upd_w2 + 2 * 64 * 64, upd_b2 + 2 * 64,
      raw, n_tiles, N);

  cg_head_kernel<<<blocks, blk, 0, stream>>>(raw, fx_w1, fx_b1, fx_w2, fx_b2, pw, pb,
                                             props, af, n_tiles, N);
}
